// DSA_32255204393142
// MI455X (gfx1250) — matrix operand order repaired and checked by reading, not compiled, not run on hardware
//
#include <hip/hip_runtime.h>
#include <hip/hip_bf16.h>
#include <math.h>

typedef __attribute__((ext_vector_type(16))) _Float16 v16h;
typedef __attribute__((ext_vector_type(8)))  float    v8f;

#define B_   2
#define S_   2048
#define D_   1024
#define H_   16
#define DK_  64
#define HI_  4
#define DI_  64
#define TOPK_ 512
#define NIDX_ 256

union VecH { v16h v; unsigned int u[8]; };
union VecF { v8f  v; float f[8]; };

__device__ __forceinline__ unsigned int packf16(float a, float b) {
    union { _Float16 h; unsigned short s; } ua, ub;
    ua.h = (_Float16)a; ub.h = (_Float16)b;
    return (unsigned int)ua.s | ((unsigned int)ub.s << 16);
}
__device__ __forceinline__ unsigned short h16(float a) {
    union { _Float16 h; unsigned short s; } u; u.h = (_Float16)a; return u.s;
}
__device__ __forceinline__ v8f wmma_f16(v16h a, v16h b, v8f c) {
    return __builtin_amdgcn_wmma_f32_16x16x32_f16(false, a, false, b, (short)0, c, false, false);
}
__device__ __forceinline__ void ld4(VecH& dst, int qi, const unsigned int* p) {
    uint4 t = *(const uint4*)p;
    dst.u[qi + 0] = t.x; dst.u[qi + 1] = t.y; dst.u[qi + 2] = t.z; dst.u[qi + 3] = t.w;
}

// ---------------------------------------------------------------------------
// Kernel 0a: convert K,V to f16 rows for gathered WMMA B-operands; zero kl_loss
// ---------------------------------------------------------------------------
__global__ void prep_kv(const float* __restrict__ K, const float* __restrict__ V,
                        unsigned short* __restrict__ Kh, unsigned short* __restrict__ Vh,
                        float* __restrict__ out)
{
    int i = blockIdx.x * blockDim.x + threadIdx.x;
    int n = B_ * S_ * DK_;
    if (i < n) { Kh[i] = h16(K[i]); Vh[i] = h16(V[i]); }
    if (i == 0) out[(size_t)B_ * S_ * H_ * DK_] = 0.f;   // kl_loss = 0
}

// ---------------------------------------------------------------------------
// Kernel 0b: one-time f16 conversions for the indexer GEMM:
//   xh   : x as f16 row-major            (A operand: contiguous K pairs)
//   Wqt/Wkt : W transposed, f16, [n][k]  (B operand: contiguous K pairs)
// ---------------------------------------------------------------------------
__global__ void prep_xw(const float* __restrict__ x,
                        const float* __restrict__ Wq, const float* __restrict__ Wk,
                        unsigned short* __restrict__ xh,
                        unsigned short* __restrict__ Wqt, unsigned short* __restrict__ Wkt)
{
    const int i = blockIdx.x * blockDim.x + threadIdx.x;
    const int nstride = gridDim.x * blockDim.x;
    for (int idx = i; idx < B_ * S_ * D_; idx += nstride) xh[idx] = h16(x[idx]);
    for (int idx = i; idx < NIDX_ * D_; idx += nstride) {
        int n = idx >> 10, k = idx & (D_ - 1);
        Wqt[idx] = h16(Wq[(size_t)k * NIDX_ + n]);
        Wkt[idx] = h16(Wk[(size_t)k * NIDX_ + n]);
    }
}

// ---------------------------------------------------------------------------
// Kernel 1: indexer projection GEMM (x @ W + b) fused with per-head LayerNorm,
// output f16. One wave = 16 rows x one head (64 cols) = 4 WMMA N-tiles.
// All operands pre-converted f16 -> pure b128 loads feeding WMMA.
// ---------------------------------------------------------------------------
__global__ void __launch_bounds__(128) idx_proj_ln(
        const unsigned short* __restrict__ xh,
        const unsigned short* __restrict__ Wqt, const unsigned short* __restrict__ bq,
        const unsigned short* __restrict__ Wkt, const unsigned short* __restrict__ bk,
        const float* __restrict__ bqf, const float* __restrict__ bkf,
        const float* __restrict__ gamma, const float* __restrict__ beta,
        unsigned short* __restrict__ qidx, unsigned short* __restrict__ kidx)
{
    (void)bq; (void)bk;
    const int wid  = threadIdx.x >> 5, lane = threadIdx.x & 31;
    const int half = lane >> 4,        lm   = lane & 15;
    const int mtile = blockIdx.x * 4 + wid;
    const int h = blockIdx.y;
    const unsigned short* Wt = blockIdx.z ? Wkt : Wqt;
    const float* bb  = blockIdx.z ? bkf : bqf;
    unsigned short* outp = blockIdx.z ? kidx : qidx;

    const int arow = mtile * 16 + lm;
    const unsigned int* xrow = (const unsigned int*)(xh + (size_t)arow * D_);
    const unsigned int* wrow[4];
    #pragma unroll
    for (int nt = 0; nt < 4; ++nt)
        wrow[nt] = (const unsigned int*)(Wt + (size_t)(h * 64 + nt * 16 + lm) * D_);

    VecF acc[4];
    for (int nt = 0; nt < 4; ++nt) for (int v = 0; v < 8; ++v) acc[nt].f[v] = 0.f;

    for (int kk = 0; kk < D_; kk += 32) {
        const int kd = kk >> 1;                 // dword index into the row
        VecH a;
        ld4(a, 0, xrow + kd + half * 4);        // K = kk + [0..7]  (this half)
        ld4(a, 4, xrow + kd + 8 + half * 4);    // K = kk + 16 + [0..7]
        #pragma unroll
        for (int nt = 0; nt < 4; ++nt) {
            VecH bm;
            ld4(bm, 0, wrow[nt] + kd + half * 4);
            ld4(bm, 4, wrow[nt] + kd + 8 + half * 4);
            acc[nt].v = wmma_f16(a.v, bm.v, acc[nt].v);
        }
    }
    // bias + LayerNorm over this head's 64 outputs (per output row)
    float s1[8], s2[8];
    for (int v = 0; v < 8; ++v) { s1[v] = 0.f; s2[v] = 0.f; }
    #pragma unroll
    for (int nt = 0; nt < 4; ++nt) {
        float bv = bb[h * 64 + nt * 16 + lm];
        #pragma unroll
        for (int v = 0; v < 8; ++v) {
            float val = acc[nt].f[v] + bv;
            acc[nt].f[v] = val;
            s1[v] += val; s2[v] += val * val;
        }
    }
    #pragma unroll
    for (int m = 1; m < 16; m <<= 1) {
        #pragma unroll
        for (int v = 0; v < 8; ++v) {
            s1[v] += __shfl_xor(s1[v], m, 32);
            s2[v] += __shfl_xor(s2[v], m, 32);
        }
    }
    float mean[8], rstd[8];
    #pragma unroll
    for (int v = 0; v < 8; ++v) {
        mean[v] = s1[v] * (1.f / 64.f);
        float var = s2[v] * (1.f / 64.f) - mean[v] * mean[v];
        rstd[v] = rsqrtf(var + 1e-5f);
    }
    #pragma unroll
    for (int nt = 0; nt < 4; ++nt) {
        int nl = nt * 16 + lm;
        float g = gamma[nl], be = beta[nl];
        #pragma unroll
        for (int v = 0; v < 8; ++v) {
            int rout = mtile * 16 + v + 8 * half;
            float y = (acc[nt].f[v] - mean[v]) * rstd[v] * g + be;
            outp[((size_t)rout * HI_ + h) * DI_ + nl] = h16(y);
        }
    }
}

// ---------------------------------------------------------------------------
// Kernel 2: index_scores[b,s,t] = sum_h w[h] * relu(k_idx[b,s,h,:].q_idx[b,t,h,:])
// One wave computes a 16x16 (s,t) tile; 2 WMMA per head (K=64); b128 loads.
// ---------------------------------------------------------------------------
__global__ void __launch_bounds__(128) idx_scores(
        const unsigned short* __restrict__ kidx,
        const unsigned short* __restrict__ qidx,
        const float* __restrict__ wh,
        float* __restrict__ scores)
{
    const int wid  = threadIdx.x >> 5, lane = threadIdx.x & 31;
    const int half = lane >> 4,        lm   = lane & 15;
    const int stile = blockIdx.x * 4 + wid;
    const int ttile = blockIdx.y;
    const int b = blockIdx.z;

    const int srow = stile * 16 + lm;
    const int trow = ttile * 16 + lm;

    float tot[8];
    for (int v = 0; v < 8; ++v) tot[v] = 0.f;

    #pragma unroll
    for (int h = 0; h < HI_; ++h) {
        const unsigned int* ka = (const unsigned int*)(kidx + (((size_t)(b * S_ + srow)) * HI_ + h) * DI_);
        const unsigned int* qb = (const unsigned int*)(qidx + (((size_t)(b * S_ + trow)) * HI_ + h) * DI_);
        VecH alo, ahi, blo, bhi;
        ld4(alo, 0, ka + half * 4);        ld4(alo, 4, ka + 8 + half * 4);
        ld4(ahi, 0, ka + 16 + half * 4);   ld4(ahi, 4, ka + 24 + half * 4);
        ld4(blo, 0, qb + half * 4);        ld4(blo, 4, qb + 8 + half * 4);
        ld4(bhi, 0, qb + 16 + half * 4);   ld4(bhi, 4, qb + 24 + half * 4);
        v8f z = {0.f,0.f,0.f,0.f,0.f,0.f,0.f,0.f};
        VecF c;
        c.v = wmma_f16(ahi.v, bhi.v, z);
        c.v = wmma_f16(alo.v, blo.v, c.v);
        float w = wh[h];
        #pragma unroll
        for (int v = 0; v < 8; ++v) tot[v] += w * fmaxf(c.f[v], 0.f);
    }
    #pragma unroll
    for (int v = 0; v < 8; ++v) {
        int s = stile * 16 + v + 8 * half;
        int t = ttile * 16 + lm;
        scores[((size_t)(b * S_ + s)) * S_ + t] = tot[v];
    }
}

// ---------------------------------------------------------------------------
// Kernel 3: exact top-512 of 2048 per row. MSB-first 8-bit radix select on
// monotone uint keys, then deterministic scan-based compaction (ties at the
// threshold take the smallest indices, matching jax.lax.top_k).
// ---------------------------------------------------------------------------
__global__ void __launch_bounds__(256) topk_rows(
        const float* __restrict__ scores, int* __restrict__ topidx)
{
    __shared__ unsigned int keys[S_];
    __shared__ unsigned int hist[256];
    __shared__ unsigned int cg[256], ce[256];
    __shared__ unsigned int shPrefix, shRemain;

    const int tid = threadIdx.x;
    const int s = blockIdx.x, b = blockIdx.y;
    const float* row = scores + ((size_t)(b * S_ + s)) * S_;
    int* outp = topidx + ((size_t)(b * S_ + s)) * TOPK_;

    for (int i = tid; i < S_; i += 256) {
        union { float f; unsigned int u; } cv; cv.f = row[i];
        unsigned int u = cv.u;
        keys[i] = (u & 0x80000000u) ? ~u : (u | 0x80000000u);
    }
    if (tid == 0) { shPrefix = 0u; shRemain = TOPK_; }
    __syncthreads();

    for (int pass = 0; pass < 4; ++pass) {
        const int shift = 24 - 8 * pass;
        const unsigned int maskHi = (pass == 0) ? 0u : (0xFFFFFFFFu << (shift + 8));
        if (tid < 256) hist[tid] = 0u;
        __syncthreads();
        const unsigned int pref = shPrefix;
        for (int i = tid; i < S_; i += 256) {
            unsigned int k = keys[i];
            if ((k & maskHi) == pref) atomicAdd(&hist[(k >> shift) & 0xFFu], 1u);
        }
        __syncthreads();
        if (tid == 0) {
            unsigned int cum = 0, remain = shRemain, chosen = 0;
            for (int bin = 255; bin >= 0; --bin) {
                unsigned int c = hist[bin];
                if (cum + c >= remain) { chosen = (unsigned int)bin; break; }
                cum += c;
            }
            shPrefix = pref | (chosen << shift);
            shRemain = remain - cum;
        }
        __syncthreads();
    }
    const unsigned int T = shPrefix;
    const unsigned int needEq = shRemain;
    const int nGt = TOPK_ - (int)needEq;

    const int base = tid * 8;
    unsigned int cgt = 0, ceq = 0;
    for (int e = 0; e < 8; ++e) {
        unsigned int k = keys[base + e];
        if (k > T) ++cgt; else if (k == T) ++ceq;
    }
    cg[tid] = cgt; ce[tid] = ceq;
    __syncthreads();
    if (tid == 0) {
        unsigned int rg = 0, re = 0;
        for (int t = 0; t < 256; ++t) {
            unsigned int a = cg[t], c = ce[t];
            cg[t] = rg; ce[t] = re;
            rg += a; re += c;
        }
    }
    __syncthreads();
    unsigned int pg = cg[tid], pe = ce[tid];
    for (int e = 0; e < 8; ++e) {
        int i = base + e;
        unsigned int k = keys[i];
        if (k > T) { outp[pg++] = i; }
        else if (k == T) { if (pe < needEq) outp[nGt + (int)pe] = i; ++pe; }
    }
}

// ---------------------------------------------------------------------------
// Kernel 4: gathered flash attention. One block per (b,s); 4 waves x 128 keys.
// scores: A = Q (16 heads x 64), B = gathered K^T via row-major f16 rows.
// P C-layout -> A-layout via per-wave LDS staging (s_wait_dscnt 0 guards).
// ---------------------------------------------------------------------------
__global__ void __launch_bounds__(128) sparse_attn(
        const float* __restrict__ Q,
        const unsigned short* __restrict__ Kh,
        const unsigned short* __restrict__ Vh,
        const int* __restrict__ topidx,
        float* __restrict__ out)
{
    __shared__ float OL[4][16][64];
    __shared__ float mL[4][16], lL[4][16];
    __shared__ unsigned short Pls[4][16][32];
    __shared__ int idxL[4][32];

    const int wid  = threadIdx.x >> 5, lane = threadIdx.x & 31;
    const int half = lane >> 4,        lm   = lane & 15;
    const int s = blockIdx.x, b = blockIdx.y;

    // A = Q tile: rows = 16 heads, K = DK=64 in two 32-wide chunks
    const float* qrow = Q + (((size_t)(b * H_ + lm)) * S_ + s) * DK_;
    VecH aq0, aq1;
    #pragma unroll
    for (int j = 0; j < 8; ++j) {
        int koff = half * 16 + j * 2;
        aq0.u[j] = packf16(qrow[koff],      qrow[koff + 1]);
        aq1.u[j] = packf16(qrow[koff + 32], qrow[koff + 33]);
    }

    const int* ridx = topidx + ((size_t)(b * S_ + s)) * TOPK_;
    const unsigned int*   Kru = (const unsigned int*)(Kh + (size_t)b * S_ * DK_);
    const unsigned short* Vr  = Vh + (size_t)b * S_ * DK_;

    float mrun[8], lrun[8];
    VecF accO[4];
    for (int v = 0; v < 8; ++v) { mrun[v] = -3.0e38f; lrun[v] = 0.f; }
    for (int nt = 0; nt < 4; ++nt) for (int v = 0; v < 8; ++v) accO[nt].f[v] = 0.f;

    for (int cc = 0; cc < 4; ++cc) {
        const int kbase = wid * 128 + cc * 32;
        idxL[wid][lane] = ridx[kbase + lane];
        asm volatile("s_wait_dscnt 0" ::: "memory");

        const int t0 = idxL[wid][lm], t1 = idxL[wid][16 + lm];
        const unsigned int* kr0 = Kru + (size_t)t0 * 32;
        const unsigned int* kr1 = Kru + (size_t)t1 * 32;
        VecH b0l, b0h, b1l, b1h;
        ld4(b0l, 0, kr0 + half * 8);        ld4(b0l, 4, kr0 + half * 8 + 4);
        ld4(b0h, 0, kr0 + 16 + half * 8);   ld4(b0h, 4, kr0 + 20 + half * 8);
        ld4(b1l, 0, kr1 + half * 8);        ld4(b1l, 4, kr1 + half * 8 + 4);
        ld4(b1h, 0, kr1 + 16 + half * 8);   ld4(b1h, 4, kr1 + 20 + half * 8);

        v8f z = {0.f,0.f,0.f,0.f,0.f,0.f,0.f,0.f};
        VecF sc0, sc1;
        sc0.v = wmma_f16(aq1.v, b0h.v, z);
        sc0.v = wmma_f16(aq0.v, b0l.v, sc0.v);
        sc1.v = wmma_f16(aq1.v, b1h.v, z);
        sc1.v = wmma_f16(aq0.v, b1l.v, sc1.v);

        float mx[8];
        #pragma unroll
        for (int v = 0; v < 8; ++v) {
            sc0.f[v] *= 0.125f; sc1.f[v] *= 0.125f;
            mx[v] = fmaxf(sc0.f[v], sc1.f[v]);
        }
        #pragma unroll
        for (int m = 1; m < 16; m <<= 1)
            #pragma unroll
            for (int v = 0; v < 8; ++v) mx[v] = fmaxf(mx[v], __shfl_xor(mx[v], m, 32));

        float rs[8];
        #pragma unroll
        for (int v = 0; v < 8; ++v) {
            float mnew = fmaxf(mrun[v], mx[v]);
            float sf = __expf(mrun[v] - mnew);
            float p0 = __expf(sc0.f[v] - mnew);
            float p1 = __expf(sc1.f[v] - mnew);
            sc0.f[v] = p0; sc1.f[v] = p1;
            rs[v] = p0 + p1;
            lrun[v] *= sf;
            mrun[v] = mnew;
            #pragma unroll
            for (int nt = 0; nt < 4; ++nt) accO[nt].f[v] *= sf;
        }
        #pragma unroll
        for (int m = 1; m < 16; m <<= 1)
            #pragma unroll
            for (int v = 0; v < 8; ++v) rs[v] += __shfl_xor(rs[v], m, 32);
        #pragma unroll
        for (int v = 0; v < 8; ++v) lrun[v] += rs[v];

        // stage P: write C-layout, read back A-layout
        #pragma unroll
        for (int v = 0; v < 8; ++v) {
            int m = v + 8 * half;
            Pls[wid][m][lm]      = h16(sc0.f[v]);
            Pls[wid][m][16 + lm] = h16(sc1.f[v]);
        }
        asm volatile("s_wait_dscnt 0" ::: "memory");
        const unsigned int* pr = (const unsigned int*)&Pls[wid][lm][0];
        VecH pa;
        ld4(pa, 0, pr + half * 4);
        ld4(pa, 4, pr + 8 + half * 4);

        // O += P x V_gathered
        #pragma unroll
        for (int nt = 0; nt < 4; ++nt) {
            int d = nt * 16 + lm;
            VecH bv;
            #pragma unroll
            for (int j = 0; j < 8; ++j) {
                int krow = (j >> 2) * 16 + half * 8 + (j & 3) * 2;
                int r0 = idxL[wid][krow], r1 = idxL[wid][krow + 1];
                unsigned int lo = Vr[(size_t)r0 * DK_ + d];
                unsigned int hi = Vr[(size_t)r1 * DK_ + d];
                bv.u[j] = lo | (hi << 16);
            }
            accO[nt].v = wmma_f16(pa.v, bv.v, accO[nt].v);
        }
    }

    // dump wave partials, merge 4 ways, write output
    #pragma unroll
    for (int nt = 0; nt < 4; ++nt)
        #pragma unroll
        for (int v = 0; v < 8; ++v)
            OL[wid][v + 8 * half][nt * 16 + lm] = accO[nt].f[v];
    if (lm == 0) {
        #pragma unroll
        for (int v = 0; v < 8; ++v) {
            mL[wid][v + 8 * half] = mrun[v];
            lL[wid][v + 8 * half] = lrun[v];
        }
    }
    __syncthreads();

    const int tid = threadIdx.x;
    float* orow = out + ((size_t)(b * S_ + s)) * (H_ * DK_);
    #pragma unroll
    for (int e = 0; e < 8; ++e) {
        int o = tid * 8 + e;
        int h = o >> 6, d = o & 63;
        float mmax = fmaxf(fmaxf(mL[0][h], mL[1][h]), fmaxf(mL[2][h], mL[3][h]));
        float Ls = 0.f, Os = 0.f;
        #pragma unroll
        for (int w2 = 0; w2 < 4; ++w2) {
            float f = __expf(mL[w2][h] - mmax);
            Ls += f * lL[w2][h];
            Os += f * OL[w2][h][d];
        }
        orow[o] = Os / Ls;
    }
}

// ---------------------------------------------------------------------------
extern "C" void kernel_launch(void* const* d_in, const int* in_sizes, int n_in,
                              void* d_out, int out_size, void* d_ws, size_t ws_size,
                              hipStream_t stream)
{
    (void)in_sizes; (void)n_in; (void)out_size; (void)ws_size;
    const float* x     = (const float*)d_in[0];
    const float* Q     = (const float*)d_in[1];
    const float* K     = (const float*)d_in[2];
    const float* V     = (const float*)d_in[3];
    const float* Wq    = (const float*)d_in[4];
    const float* bq    = (const float*)d_in[5];
    const float* Wk    = (const float*)d_in[6];
    const float* bk    = (const float*)d_in[7];
    const float* gamma = (const float*)d_in[8];
    const float* beta  = (const float*)d_in[9];
    const float* wh    = (const float*)d_in[10];
    float* out = (float*)d_out;

    char* ws = (char*)d_ws;
    const size_t MB = 1u << 20;
    unsigned short* qidx = (unsigned short*)(ws);                 // [0, 2MB)
    unsigned short* kidx = (unsigned short*)(ws + 2 * MB);        // [2MB, 4MB)
    unsigned short* Kh   = (unsigned short*)(ws + 4 * MB);        // [4MB, 4.5MB)
    unsigned short* Vh   = (unsigned short*)(ws + 4 * MB + 512 * 1024);
    float* scores = (float*)(ws + 5 * MB);                        // [5MB, 37MB)
    int*   topidx = (int*)(ws + 37 * MB);                         // [37MB, 45MB)
    unsigned short* xh   = (unsigned short*)(ws + 45 * MB);       // [45MB, 53MB)
    unsigned short* Wqt  = (unsigned short*)(ws + 53 * MB);       // [53MB, 53.5MB)
    unsigned short* Wkt  = (unsigned short*)(ws + 53 * MB + 512 * 1024);

    prep_kv<<<dim3((B_ * S_ * DK_ + 255) / 256), dim3(256), 0, stream>>>(K, V, Kh, Vh, out);
    prep_xw<<<dim3(2048), dim3(256), 0, stream>>>(x, Wq, Wk, xh, Wqt, Wkt);
    idx_proj_ln<<<dim3(64, HI_, 2), dim3(128), 0, stream>>>(
        xh, Wqt, nullptr, Wkt, nullptr, bq, bk, gamma, beta, qidx, kidx);
    idx_scores<<<dim3(S_ / 64, S_ / 16, B_), dim3(128), 0, stream>>>(kidx, qidx, wh, scores);
    topk_rows<<<dim3(S_, B_), dim3(256), 0, stream>>>(scores, topidx);
    sparse_attn<<<dim3(S_, B_), dim3(128), 0, stream>>>(Q, Kh, Vh, topidx, out);
}
